// IPFeatExtractor_91268055040126
// MI455X (gfx1250) — matrix operand order repaired and checked by reading, not compiled, not run on hardware
//
#include <hip/hip_runtime.h>

#define TT 16
#define NN 2048
#define MM 2048
#define IND 128
#define OUTD 128
#define HIDN 128
#define FEATN 256

typedef __attribute__((ext_vector_type(16))) __bf16 v16bf;
typedef __attribute__((ext_vector_type(8)))  __bf16 v8bf;
typedef __attribute__((ext_vector_type(2)))  __bf16 v2bf;
typedef __attribute__((ext_vector_type(8)))  float  v8f;
typedef __attribute__((ext_vector_type(4)))  int    v4i;

struct bfpair { v8bf lo, hi; };
struct pk8 { v2bf p[8]; };  // 32 bytes == v16bf
struct pk4 { v2bf p[4]; };  // 16 bytes == v8bf

static __device__ inline unsigned lane_id() { return threadIdx.x & 31u; }

static __device__ inline v2bf pk_bf16(float lo, float hi) {
#if __has_builtin(__builtin_amdgcn_cvt_pk_bf16_f32)
  return __builtin_amdgcn_cvt_pk_bf16_f32(lo, hi);
#else
  v2bf r;
  r[0] = (__bf16)lo;
  r[1] = (__bf16)hi;
  return r;
#endif
}
static __device__ inline __bf16 f2bf(float f) { return (__bf16)f; }
static __device__ inline float bf2f(__bf16 b) { return (float)b; }

static __device__ inline v8f v8f_zero() {
  v8f z = {0.f,0.f,0.f,0.f,0.f,0.f,0.f,0.f};
  return z;
}
static __device__ inline float sigm(float x) { return 1.f / (1.f + __expf(-x)); }

static __device__ inline v8f wmma_bf16(v16bf a, v16bf b, v8f c) {
  return __builtin_amdgcn_wmma_f32_16x16x32_bf16(false, a, false, b, (short)0, c, false, false);
}

// ---- CDNA5 async VMEM->LDS path (ASYNCcnt), with register-bounce fallback ----
#if __has_builtin(__builtin_amdgcn_global_load_async_to_lds_b128) && \
    __has_builtin(__builtin_amdgcn_s_wait_asynccnt)
#define ASYNC_LDS 1
typedef __attribute__((address_space(1))) v4i as1_v4i;
typedef __attribute__((address_space(3))) v4i as3_v4i;
static __device__ inline void async_copy16(const __bf16* gsrc, __bf16* ldst) {
  __builtin_amdgcn_global_load_async_to_lds_b128((as1_v4i*)gsrc, (as3_v4i*)ldst, 0, 0);
}
static __device__ inline void async_wait0() { __builtin_amdgcn_s_wait_asynccnt(0); }
#endif

// pack 16 f32 (from four float4) into a v16bf operand with hw pack-converts
static __device__ inline v16bf cvt_f32x16(float4 f0, float4 f1, float4 f2, float4 f3) {
  pk8 t;
  t.p[0] = pk_bf16(f0.x, f0.y); t.p[1] = pk_bf16(f0.z, f0.w);
  t.p[2] = pk_bf16(f1.x, f1.y); t.p[3] = pk_bf16(f1.z, f1.w);
  t.p[4] = pk_bf16(f2.x, f2.y); t.p[5] = pk_bf16(f2.z, f2.w);
  t.p[6] = pk_bf16(f3.x, f3.y); t.p[7] = pk_bf16(f3.z, f3.w);
  return __builtin_bit_cast(v16bf, t);
}

// ---- A-fragment 16x32 from row-major bf16: two contiguous 16B loads per lane
static __device__ inline v16bf load_a_bf16(const __bf16* src, long row0, int k0, int ld) {
  unsigned l = lane_id();
  long r = row0 + (l & 15u);
  int kb = k0 + (int)((l >> 4) << 3);
  const __bf16* p = src + r * (long)ld + kb;
  bfpair parts;
  parts.lo = *(const v8bf*)p;          // K = kb .. kb+7
  parts.hi = *(const v8bf*)(p + 16);   // K = kb+16 .. kb+23
  return __builtin_bit_cast(v16bf, parts);
}
// ---- A-fragment 16x32 from row-major fp32: four float4 loads + pack-convert
static __device__ inline v16bf load_a_f32(const float* src, long row0, int k0, int ld) {
  unsigned l = lane_id();
  long r = row0 + (l & 15u);
  int kb = k0 + (int)((l >> 4) << 3);
  const float4* p = (const float4*)(src + r * (long)ld + kb);
  const float4* q = (const float4*)(src + r * (long)ld + kb + 16);
  return cvt_f32x16(p[0], p[1], q[0], q[1]);
}
// ---- B-fragment 32x16 from swizzled layout sw[((k>>4)*ld + n)*16 + (k&15)]
static __device__ inline v16bf load_b_swz(const __bf16* sw, long k0, int col0, int ld) {
  unsigned l = lane_id();
  int c = col0 + (int)(l & 15u);
  long kt = (k0 >> 4);
  const __bf16* p = sw + (kt * (long)ld + c) * 16 + (long)((l >> 4) << 3);
  bfpair parts;
  parts.lo = *(const v8bf*)p;
  parts.hi = *(const v8bf*)(p + (long)ld * 16);
  return __builtin_bit_cast(v16bf, parts);
}
// ---- B-fragment from an LDS image of one 32-K swizzled chunk (2 ktiles x ld cols)
static __device__ inline v16bf load_b_lds(const __bf16* sh, int col0, int ld) {
  unsigned l = lane_id();
  int c = col0 + (int)(l & 15u);
  const __bf16* p = sh + (long)c * 16 + (long)((l >> 4) << 3);
  bfpair parts;
  parts.lo = *(const v8bf*)p;
  parts.hi = *(const v8bf*)(p + (long)ld * 16);
  return __builtin_bit_cast(v16bf, parts);
}
// ---- Store a 16x16 f32 D tile into B-swizzled bf16 layout: one 16B store per lane
static __device__ inline void store_tile_bswz(__bf16* sw, long row0, int col0, int ld, v8f acc) {
  unsigned l = lane_id();
  int c = col0 + (int)(l & 15u);
  long kt = row0 >> 4;  // row0 is a multiple of 16
  __bf16* p = sw + (kt * (long)ld + c) * 16 + (long)((l >> 4) << 3);
  pk4 t;
  t.p[0] = pk_bf16(acc[0], acc[1]);
  t.p[1] = pk_bf16(acc[2], acc[3]);
  t.p[2] = pk_bf16(acc[4], acc[5]);
  t.p[3] = pk_bf16(acc[6], acc[7]);
  *(v8bf*)p = __builtin_bit_cast(v8bf, t);
}

// ---------------- prep: bf16 conversions into swizzled B layouts, state init ----------------
__global__ void k_prep(const float* W1, const float* W2, const float* Wg,
                       const float* Wih, const float* Whh,
                       __bf16* W1b, __bf16* W2b, __bf16* Wgb,
                       __bf16* WihT, __bf16* WhhT,
                       __bf16* hA, float* cst) {
  int tid = blockIdx.x * blockDim.x + threadIdx.x;
  int stride = gridDim.x * blockDim.x;
  for (int i = tid; i < IND * OUTD; i += stride) {      // (k,n) of [128,128]
    int k = i / OUTD, n = i % OUTD;
    long sw = ((long)(k >> 4) * OUTD + n) * 16 + (k & 15);
    W1b[sw] = f2bf(W1[i]);
    W2b[sw] = f2bf(W2[i]);
  }
  for (int i = tid; i < FEATN * FEATN; i += stride) {   // (k,n) of [256,256]
    int k = i / FEATN, n = i % FEATN;
    Wgb[((long)(k >> 4) * FEATN + n) * 16 + (k & 15)] = f2bf(Wg[i]);
  }
  for (int i = tid; i < 4 * HIDN * FEATN; i += stride) { // Wih[r,f] -> B(k=f, n=r)
    int r = i / FEATN, f = i % FEATN;
    WihT[((long)(f >> 4) * (4 * HIDN) + r) * 16 + (f & 15)] = f2bf(Wih[i]);
  }
  for (int i = tid; i < 4 * HIDN * HIDN; i += stride) {  // Whh[r,f] -> B(k=f, n=r)
    int r = i / HIDN, f = i % HIDN;
    WhhT[((long)(f >> 4) * (4 * HIDN) + r) * 16 + (f & 15)] = f2bf(Whh[i]);
  }
  for (int i = tid; i < MM * HIDN; i += stride) { hA[i] = f2bf(0.f); cst[i] = 0.f; }
}

// ---------------- stage 1: xw = x@W (bf16 WMMA), output B-swizzled ----------------
__global__ void __launch_bounds__(256) k_xw(const float* x, const __bf16* W1b, const __bf16* W2b,
                                            __bf16* xw1, __bf16* xw2) {
  int wave = threadIdx.x >> 5;
  long row0 = (long)blockIdx.x * 128 + wave * 16;  // rows of [T*N, 128]
  for (int w = 0; w < 2; ++w) {
    const __bf16* Wb = w ? W2b : W1b;
    __bf16* out = w ? xw2 : xw1;
    v8f acc[8];
#pragma unroll
    for (int ct = 0; ct < 8; ++ct) acc[ct] = v8f_zero();
    for (int k0 = 0; k0 < IND; k0 += 32) {
      v16bf a = load_a_f32(x, row0, k0, IND);
#pragma unroll
      for (int ct = 0; ct < 8; ++ct)
        acc[ct] = wmma_bf16(a, load_b_swz(Wb, k0, ct * 16, OUTD), acc[ct]);
    }
#pragma unroll
    for (int ct = 0; ct < 8; ++ct) store_tile_bswz(out, row0, ct * 16, OUTD, acc[ct]);
  }
}

// ---------------- stage 2: h = A[t] @ xw + bias -> feat half (row-major bf16) ----------------
// B panel staged through LDS (async VMEM->LDS when available), double-buffered.
__global__ void __launch_bounds__(256) k_spmm(const float* Ain, const float* Aout,
                                              const __bf16* xw1, const __bf16* xw2,
                                              const float* b1, const float* b2,
                                              __bf16* feat) {
  __shared__ alignas(32) __bf16 Bsh[2][2 * OUTD * 16];  // 2 x 8KB
  int t = blockIdx.z;
  int half = blockIdx.y;
  int wave = threadIdx.x >> 5;
  unsigned l = lane_id();
  long i0 = (long)blockIdx.x * 128 + wave * 16;
  const float* Aseg = (half ? Aout : Ain) + (long)t * NN * NN;
  const __bf16* Bsw = (half ? xw2 : xw1) + (long)t * NN * OUTD;  // swizzled segment
  const float* bias = half ? b2 : b1;
  long r = i0 + (long)(l & 15u);
  int kb = (int)((l >> 4) << 3);
  v8f acc[8];
#pragma unroll
  for (int ct = 0; ct < 8; ++ct) acc[ct] = v8f_zero();

#ifdef ASYNC_LDS
  {
    const __bf16* src = Bsw + (long)threadIdx.x * 16;
    async_copy16(src, &Bsh[0][threadIdx.x * 16]);
    async_copy16(src + 8, &Bsh[0][threadIdx.x * 16 + 8]);
    async_wait0();
  }
  __syncthreads();
  for (int k0 = 0; k0 < NN; k0 += 32) {
    int cur = (k0 >> 5) & 1;
    bool have = (k0 + 32) < NN;
    if (have) {
      const __bf16* src = Bsw + ((long)((k0 + 32) >> 4) * OUTD) * 16 + (long)threadIdx.x * 16;
      async_copy16(src, &Bsh[cur ^ 1][threadIdx.x * 16]);
      async_copy16(src + 8, &Bsh[cur ^ 1][threadIdx.x * 16 + 8]);
    }
    __builtin_prefetch(Aseg + r * NN + k0 + kb + 64, 0, 1);
    v16bf a = load_a_f32(Aseg, i0, k0, NN);
#pragma unroll
    for (int ct = 0; ct < 8; ++ct)
      acc[ct] = wmma_bf16(a, load_b_lds(&Bsh[cur][0], ct * 16, OUTD), acc[ct]);
    if (have) async_wait0();
    __syncthreads();
  }
#else
  {
    v16bf v = *(const v16bf*)(Bsw + (long)threadIdx.x * 16);
    *(v16bf*)(&Bsh[0][threadIdx.x * 16]) = v;
  }
  __syncthreads();
  for (int k0 = 0; k0 < NN; k0 += 32) {
    int cur = (k0 >> 5) & 1;
    bool have = (k0 + 32) < NN;
    v16bf nxt;
    if (have)
      nxt = *(const v16bf*)(Bsw + ((long)((k0 + 32) >> 4) * OUTD) * 16 + (long)threadIdx.x * 16);
    __builtin_prefetch(Aseg + r * NN + k0 + kb + 64, 0, 1);
    v16bf a = load_a_f32(Aseg, i0, k0, NN);
#pragma unroll
    for (int ct = 0; ct < 8; ++ct)
      acc[ct] = wmma_bf16(a, load_b_lds(&Bsh[cur][0], ct * 16, OUTD), acc[ct]);
    if (have) *(v16bf*)(&Bsh[cur ^ 1][threadIdx.x * 16]) = nxt;
    __syncthreads();
  }
#endif

#pragma unroll
  for (int ct = 0; ct < 8; ++ct) {
    int col = ct * 16 + (int)(l & 15u);
    float bv = bias[col];
#pragma unroll
    for (int e = 0; e < 8; ++e) {
      long row = i0 + (long)((l >> 4) * 8 + e);
      feat[((long)t * NN + row) * FEATN + half * OUTD + col] = f2bf(acc[ct][e] + bv);
    }
  }
}

// ---------------- stage 3: g = feat @ Wg (WMMA), output B-swizzled ----------------
__global__ void __launch_bounds__(256) k_g(const __bf16* feat, const __bf16* Wgb, __bf16* g) {
  int t = blockIdx.z;
  int half = blockIdx.y;
  int wave = threadIdx.x >> 5;
  long i0 = (long)blockIdx.x * 128 + wave * 16;
  const __bf16* Aseg = feat + (long)t * NN * FEATN;
  __bf16* gsw = g + (long)t * NN * FEATN;
  v8f acc[8];
#pragma unroll
  for (int ct = 0; ct < 8; ++ct) acc[ct] = v8f_zero();
  for (int k0 = 0; k0 < FEATN; k0 += 32) {
    v16bf a = load_a_bf16(Aseg, i0, k0, FEATN);
#pragma unroll
    for (int ct = 0; ct < 8; ++ct)
      acc[ct] = wmma_bf16(a, load_b_swz(Wgb, k0, half * 128 + ct * 16, FEATN), acc[ct]);
  }
#pragma unroll
  for (int ct = 0; ct < 8; ++ct)
    store_tile_bswz(gsw, i0, half * 128 + ct * 16, FEATN, acc[ct]);
}

// ---------------- stage 3b: s = g.a_src, d = g.a_dst (reads swizzled g) ----------------
__global__ void k_sd(const __bf16* g, const float* a_src, const float* a_dst,
                     float* s, float* d) {
  long idx = (long)blockIdx.x * blockDim.x + threadIdx.x;  // t*N + i
  long t = idx / NN;
  long i = idx % NN;
  const __bf16* base = g + t * NN * FEATN + ((i >> 4) * FEATN) * 16 + (i & 15);
  float as = 0.f, ad = 0.f;
#pragma unroll 8
  for (int f = 0; f < FEATN; ++f) {
    float v = bf2f(base[(long)f * 16]);
    as += v * a_src[f];
    ad += v * a_dst[f];
  }
  s[idx] = as;
  d[idx] = ad;
}

// ---------------- stage 4a: online softmax stats per row (float4 loads) ----------------
__global__ void __launch_bounds__(256) k_stats(const float* A, const float* s, const float* d,
                                               float* rmax, float* rden) {
  int wave = threadIdx.x >> 5;
  int lane = threadIdx.x & 31;
  long row = (long)blockIdx.x * 8 + wave;  // t*N + i
  int t = (int)(row / NN);
  long i = row % NN;
  const float* Arow = A + (long)t * NN * NN + i * NN;
  const float* drow = d + (long)t * NN;
  float si = s[row];
  float m = -1e30f, sum = 0.f;
  for (int j = lane * 4; j < NN; j += 128) {
    float4 av = *(const float4*)(Arow + j);
    float4 dv = *(const float4*)(drow + j);
    float aa[4] = {av.x, av.y, av.z, av.w};
    float dd[4] = {dv.x, dv.y, dv.z, dv.w};
#pragma unroll
    for (int q = 0; q < 4; ++q) {
      float e;
      if (aa[q] > 0.f) {
        float v = si + dd[q];
        e = v > 0.f ? v : 0.2f * v;
      } else {
        e = -1e9f;
      }
      if (e > m) { sum *= __expf(m - e); m = e; }
      sum += __expf(e - m);
    }
  }
#pragma unroll
  for (int o = 16; o > 0; o >>= 1) {
    float m2 = __shfl_xor(m, o, 32);
    float s2 = __shfl_xor(sum, o, 32);
    float mm = fmaxf(m, m2);
    sum = sum * __expf(m - mm) + s2 * __expf(m2 - mm);
    m = mm;
  }
  if (lane == 0) { rmax[row] = m; rden[row] = fmaxf(sum, 1e-20f); }
}

// ---------------- stage 4b: gat = softmax(e) @ g; alpha in-register, g staged via LDS ----------------
__global__ void __launch_bounds__(256) k_gat(const float* A, const float* s, const float* d,
                                             const float* rmax, const float* rden,
                                             const __bf16* g, __bf16* gat) {
  __shared__ alignas(32) __bf16 Bsh[2][2 * FEATN * 16];  // 2 x 16KB
  __shared__ alignas(16) float dsh[NN];                  // 8KB: d row for this t
  int t = blockIdx.z;
  int wave = threadIdx.x >> 5;
  unsigned l = lane_id();
  long i0 = (long)blockIdx.x * 128 + wave * 16;
  long r = i0 + (long)(l & 15u);  // this lane's A-fragment row
  const float* Aseg = A + (long)t * NN * NN;
  const float* drow = d + (long)t * NN;
  const __bf16* gsw = g + (long)t * NN * FEATN;  // swizzled
  float si = s[(long)t * NN + r];
  float mi = rmax[(long)t * NN + r];
  int kb = (int)((l >> 4) << 3);
  v8f acc[16];
#pragma unroll
  for (int ct = 0; ct < 16; ++ct) acc[ct] = v8f_zero();

  // stage d row + chunk 0 of g
  {
    int j = threadIdx.x * 8;
    *(float4*)(dsh + j) = *(const float4*)(drow + j);
    *(float4*)(dsh + j + 4) = *(const float4*)(drow + j + 4);
#ifdef ASYNC_LDS
    const __bf16* src = gsw + (long)threadIdx.x * 16;
    async_copy16(src, &Bsh[0][threadIdx.x * 16]);
    async_copy16(src + 8, &Bsh[0][threadIdx.x * 16 + 8]);
    const __bf16* src2 = gsw + (long)(threadIdx.x + 256) * 16;
    async_copy16(src2, &Bsh[0][(threadIdx.x + 256) * 16]);
    async_copy16(src2 + 8, &Bsh[0][(threadIdx.x + 256) * 16 + 8]);
    async_wait0();
#else
    v16bf v0 = *(const v16bf*)(gsw + (long)threadIdx.x * 16);
    v16bf v1 = *(const v16bf*)(gsw + (long)(threadIdx.x + 256) * 16);
    *(v16bf*)(&Bsh[0][threadIdx.x * 16]) = v0;
    *(v16bf*)(&Bsh[0][(threadIdx.x + 256) * 16]) = v1;
#endif
  }
  __syncthreads();

  for (int k0 = 0; k0 < NN; k0 += 32) {
    int cur = (k0 >> 5) & 1;
    bool have = (k0 + 32) < NN;
#ifdef ASYNC_LDS
    if (have) {
      const __bf16* src = gsw + ((long)((k0 + 32) >> 4) * FEATN) * 16 + (long)threadIdx.x * 16;
      async_copy16(src, &Bsh[cur ^ 1][threadIdx.x * 16]);
      async_copy16(src + 8, &Bsh[cur ^ 1][threadIdx.x * 16 + 8]);
      const __bf16* src2 = src + 256 * 16;
      async_copy16(src2, &Bsh[cur ^ 1][(threadIdx.x + 256) * 16]);
      async_copy16(src2 + 8, &Bsh[cur ^ 1][(threadIdx.x + 256) * 16 + 8]);
    }
#else
    v16bf n0v, n1v;
    if (have) {
      const __bf16* src = gsw + ((long)((k0 + 32) >> 4) * FEATN) * 16;
      n0v = *(const v16bf*)(src + (long)threadIdx.x * 16);
      n1v = *(const v16bf*)(src + (long)(threadIdx.x + 256) * 16);
    }
#endif
    int j0 = k0 + kb;
    __builtin_prefetch(Aseg + r * NN + j0 + 64, 0, 1);
    const float4* pA = (const float4*)(Aseg + r * NN + j0);
    const float4* qA = (const float4*)(Aseg + r * NN + j0 + 16);
    float4 A0 = pA[0], A1 = pA[1], A2 = qA[0], A3 = qA[1];
    float4 D0 = *(const float4*)(dsh + j0);
    float4 D1 = *(const float4*)(dsh + j0 + 4);
    float4 D2 = *(const float4*)(dsh + j0 + 16);
    float4 D3 = *(const float4*)(dsh + j0 + 20);
    float aw[16] = {A0.x,A0.y,A0.z,A0.w, A1.x,A1.y,A1.z,A1.w,
                    A2.x,A2.y,A2.z,A2.w, A3.x,A3.y,A3.z,A3.w};
    float dw[16] = {D0.x,D0.y,D0.z,D0.w, D1.x,D1.y,D1.z,D1.w,
                    D2.x,D2.y,D2.z,D2.w, D3.x,D3.y,D3.z,D3.w};
    float al[16];
#pragma unroll
    for (int e = 0; e < 16; ++e) {
      float v = si + dw[e];
      v = v > 0.f ? v : 0.2f * v;
      al[e] = (aw[e] > 0.f) ? __expf(v - mi) : 0.f;
    }
    pk8 tpk;
#pragma unroll
    for (int e = 0; e < 8; ++e) tpk.p[e] = pk_bf16(al[2 * e], al[2 * e + 1]);
    v16bf afrag = __builtin_bit_cast(v16bf, tpk);
#pragma unroll
    for (int ct = 0; ct < 16; ++ct)
      acc[ct] = wmma_bf16(afrag, load_b_lds(&Bsh[cur][0], ct * 16, FEATN), acc[ct]);
#ifdef ASYNC_LDS
    if (have) async_wait0();
#else
    if (have) {
      *(v16bf*)(&Bsh[cur ^ 1][threadIdx.x * 16]) = n0v;
      *(v16bf*)(&Bsh[cur ^ 1][(threadIdx.x + 256) * 16]) = n1v;
    }
#endif
    __syncthreads();
  }
#pragma unroll
  for (int ct = 0; ct < 16; ++ct) {
    int col = ct * 16 + (int)(l & 15u);
#pragma unroll
    for (int e = 0; e < 8; ++e) {
      long row = i0 + (long)((l >> 4) * 8 + e);
      float den = rden[(long)t * NN + row];
      gat[((long)t * NN + row) * FEATN + col] = f2bf(acc[ct][e] / den);
    }
  }
}

// ---------------- stage 5: searchsorted gather -> seqs (row-major for LSTM A-frags) ----------------
__global__ void k_gather(const int* ips, const int* cur, const __bf16* gat, __bf16* seqs) {
  long idx = (long)blockIdx.x * blockDim.x + threadIdx.x;  // t*M + m
  int t = (int)(idx / MM);
  int m = (int)(idx % MM);
  const int* ip = ips + (long)t * NN;
  int key = cur[m];
  int lo = 0, hi = NN;
  while (lo < hi) {
    int mid = (lo + hi) >> 1;
    if (ip[mid] < key) lo = mid + 1; else hi = mid;
  }
  int pos = lo < NN ? lo : NN - 1;
  bool found = (ip[pos] == key);
  const uint4* src = (const uint4*)(gat + ((long)t * NN + pos) * FEATN);
  uint4* dst = (uint4*)(seqs + idx * FEATN);
  uint4 z = {0u, 0u, 0u, 0u};
#pragma unroll
  for (int q = 0; q < FEATN * 2 / 16; ++q) dst[q] = found ? src[q] : z;
}

// ---------------- stage 6: one LSTM step (WMMA over x and h) ----------------
__global__ void __launch_bounds__(256) k_lstm(const __bf16* xt, const __bf16* hin,
                                              const __bf16* WihT, const __bf16* WhhT,
                                              const float* bih, const float* bhh,
                                              float* cst, __bf16* hout, float* hf) {
  int wave = threadIdx.x >> 5;
  unsigned l = lane_id();
  long row0 = (long)blockIdx.x * 128 + wave * 16;
  int n0 = blockIdx.y * 16;
  v8f zi = v8f_zero(), zf = v8f_zero(), zg = v8f_zero(), zo = v8f_zero();
  for (int k0 = 0; k0 < FEATN; k0 += 32) {
    v16bf a = load_a_bf16(xt, row0, k0, FEATN);
    zi = wmma_bf16(a, load_b_swz(WihT, k0, n0, 4 * HIDN), zi);
    zf = wmma_bf16(a, load_b_swz(WihT, k0, n0 + HIDN, 4 * HIDN), zf);
    zg = wmma_bf16(a, load_b_swz(WihT, k0, n0 + 2 * HIDN, 4 * HIDN), zg);
    zo = wmma_bf16(a, load_b_swz(WihT, k0, n0 + 3 * HIDN, 4 * HIDN), zo);
  }
  for (int k0 = 0; k0 < HIDN; k0 += 32) {
    v16bf a = load_a_bf16(hin, row0, k0, HIDN);
    zi = wmma_bf16(a, load_b_swz(WhhT, k0, n0, 4 * HIDN), zi);
    zf = wmma_bf16(a, load_b_swz(WhhT, k0, n0 + HIDN, 4 * HIDN), zf);
    zg = wmma_bf16(a, load_b_swz(WhhT, k0, n0 + 2 * HIDN, 4 * HIDN), zg);
    zo = wmma_bf16(a, load_b_swz(WhhT, k0, n0 + 3 * HIDN, 4 * HIDN), zo);
  }
  int col = n0 + (int)(l & 15u);
  float bi = bih[col] + bhh[col];
  float bf = bih[col + HIDN] + bhh[col + HIDN];
  float bg = bih[col + 2 * HIDN] + bhh[col + 2 * HIDN];
  float bo = bih[col + 3 * HIDN] + bhh[col + 3 * HIDN];
#pragma unroll
  for (int e = 0; e < 8; ++e) {
    long row = row0 + (long)((l >> 4) * 8 + e);
    long idx = row * HIDN + col;
    float vi = sigm(zi[e] + bi);
    float vf = sigm(zf[e] + bf);
    float vg = tanhf(zg[e] + bg);
    float vo = sigm(zo[e] + bo);
    float cn = vf * cst[idx] + vi * vg;
    cst[idx] = cn;
    float hn = vo * tanhf(cn);
    hf[idx] = hn;
    hout[idx] = f2bf(hn);
  }
}

// ---------------------------------------------------------------------------
extern "C" void kernel_launch(void* const* d_in, const int* in_sizes, int n_in,
                              void* d_out, int out_size, void* d_ws, size_t ws_size,
                              hipStream_t stream) {
  const float* x     = (const float*)d_in[0];
  const float* Ain   = (const float*)d_in[1];
  const float* Aout  = (const float*)d_in[2];
  const float* A     = (const float*)d_in[3];
  const int*   ips   = (const int*)d_in[4];
  const int*   cur   = (const int*)d_in[5];
  const float* W1    = (const float*)d_in[6];
  const float* b1    = (const float*)d_in[7];
  const float* W2    = (const float*)d_in[8];
  const float* b2    = (const float*)d_in[9];
  const float* Wg    = (const float*)d_in[10];
  const float* a_src = (const float*)d_in[11];
  const float* a_dst = (const float*)d_in[12];
  const float* Wih   = (const float*)d_in[13];
  const float* Whh   = (const float*)d_in[14];
  const float* bih   = (const float*)d_in[15];
  const float* bhh   = (const float*)d_in[16];
  (void)in_sizes; (void)n_in; (void)out_size; (void)ws_size;

  char* ws = (char*)d_ws;
  size_t o = 0;
  auto carve = [&](size_t bytes) -> char* {
    char* p = ws + o;
    o += (bytes + 255) & ~(size_t)255;
    return p;
  };
  __bf16* xw1   = (__bf16*)carve((size_t)TT * NN * OUTD * 2);   // B-swizzled per t
  __bf16* xw2   = (__bf16*)carve((size_t)TT * NN * OUTD * 2);   // B-swizzled per t
  __bf16* feat  = (__bf16*)carve((size_t)TT * NN * FEATN * 2);  // row-major
  __bf16* g     = (__bf16*)carve((size_t)TT * NN * FEATN * 2);  // B-swizzled per t
  __bf16* gat   = (__bf16*)carve((size_t)TT * NN * FEATN * 2);  // row-major
  __bf16* seqs  = (__bf16*)carve((size_t)TT * MM * FEATN * 2);  // row-major
  float*  s     = (float*)carve((size_t)TT * NN * 4);
  float*  d     = (float*)carve((size_t)TT * NN * 4);
  float*  rmax  = (float*)carve((size_t)TT * NN * 4);
  float*  rden  = (float*)carve((size_t)TT * NN * 4);
  __bf16* W1b   = (__bf16*)carve((size_t)IND * OUTD * 2);       // B-swizzled
  __bf16* W2b   = (__bf16*)carve((size_t)IND * OUTD * 2);       // B-swizzled
  __bf16* Wgb   = (__bf16*)carve((size_t)FEATN * FEATN * 2);    // B-swizzled
  __bf16* WihT  = (__bf16*)carve((size_t)FEATN * 4 * HIDN * 2); // B-swizzled
  __bf16* WhhT  = (__bf16*)carve((size_t)HIDN * 4 * HIDN * 2);  // B-swizzled
  __bf16* hA    = (__bf16*)carve((size_t)MM * HIDN * 2);
  __bf16* hB    = (__bf16*)carve((size_t)MM * HIDN * 2);
  float*  cst   = (float*)carve((size_t)MM * HIDN * 4);

  k_prep<<<256, 256, 0, stream>>>(W1, W2, Wg, Wih, Whh, W1b, W2b, Wgb, WihT, WhhT, hA, cst);

  k_xw<<<(TT * NN) / 128, 256, 0, stream>>>(x, W1b, W2b, xw1, xw2);

  k_spmm<<<dim3(NN / 128, 2, TT), 256, 0, stream>>>(Ain, Aout, xw1, xw2, b1, b2, feat);

  k_g<<<dim3(NN / 128, 2, TT), 256, 0, stream>>>(feat, Wgb, g);

  k_sd<<<(TT * NN) / 256, 256, 0, stream>>>(g, a_src, a_dst, s, d);

  k_stats<<<(TT * NN) / 8, 256, 0, stream>>>(A, s, d, rmax, rden);

  k_gat<<<dim3(NN / 128, 1, TT), 256, 0, stream>>>(A, s, d, rmax, rden, g, gat);

  k_gather<<<(TT * MM) / 256, 256, 0, stream>>>(ips, cur, gat, seqs);

  for (int t = 0; t < TT; ++t) {
    const __bf16* xt = seqs + (size_t)t * MM * FEATN;
    const __bf16* hin = (t & 1) ? hB : hA;
    __bf16* hout = (t & 1) ? hA : hB;
    k_lstm<<<dim3(MM / 128, HIDN / 16), 256, 0, stream>>>(xt, hin, WihT, WhhT, bih, bhh,
                                                          cst, hout, (float*)d_out);
  }
}
